// Attention_73005854097445
// MI455X (gfx1250) — compile-verified, matrix operand order checked by reading, not run on hardware
//
#include <hip/hip_runtime.h>
#include <math.h>

// ---------------------------------------------------------------------------
// Problem constants (reference: B=4, T=2048, C=768, H=12, D=64)
// ---------------------------------------------------------------------------
#define BB 4
#define TT 2048
#define CC 768
#define HH 12
#define DD 64
#define MM (BB * TT)            // 8192 rows for all GEMMs
#define NEG_INF_F (-1.0e9f)

typedef __attribute__((ext_vector_type(16))) __bf16 v16bf;
typedef __attribute__((ext_vector_type(8)))  float  v8f;

union Frag16 {
    uint4  q[2];   // 32 bytes = 16 bf16
    v16bf  v;
};

static __device__ __forceinline__ unsigned short f2bf(float f) {
    unsigned int u = __float_as_uint(f);
    unsigned int r = u + 0x7FFFu + ((u >> 16) & 1u);   // round-to-nearest-even
    return (unsigned short)(r >> 16);
}

// ---------------------------------------------------------------------------
// Kernel 1: fp32 -> bf16 elementwise convert
// ---------------------------------------------------------------------------
__global__ void cvt_f32_bf16(const float* __restrict__ src,
                             unsigned short* __restrict__ dst, int n) {
    int i = blockIdx.x * blockDim.x + threadIdx.x;
    if (i < n) dst[i] = f2bf(src[i]);
}

// ---------------------------------------------------------------------------
// Kernel 2: transpose 768x768 fp32 weight into bf16 WT[n][k] = W[k][n]
// ---------------------------------------------------------------------------
__global__ void transpose_w_bf16(const float* __restrict__ w,
                                 unsigned short* __restrict__ wt) {
    int i = blockIdx.x * blockDim.x + threadIdx.x;       // i = n*768 + k
    if (i >= CC * CC) return;
    int n = i / CC, k = i - n * CC;
    wt[i] = f2bf(w[k * CC + n]);
}

// ---------------------------------------------------------------------------
// Kernel 3: bf16 WMMA GEMM  C[8192,768] = A[8192,768] * Bt[768,768]^T (f32 out)
//   block = 128 threads (4 waves); each wave owns a 32x64 tile (2 A-row frags
//   x 4 B-col frags -> 8 WMMAs per 32-deep chunk); grid(M/128, N/64) = (64,12).
//   Two-stage software pipeline with a NON-unrolled outer loop: bounded live
//   set (~180 VGPRs: 64 acc + 2x48 fragment buffers) => no scratch spills,
//   while the next chunk's b128 loads issue ahead of the current WMMAs.
//   WMMA 16-bit A/B VGPR layout: lane = row/col (mod 16); half = lane/16;
//   q[0] holds K = k0 + half*8 .. +7; q[1] holds K = k0+16+half*8 .. +7.
// ---------------------------------------------------------------------------
__global__ void __launch_bounds__(128)
gemm_bf16_wmma(const unsigned short* __restrict__ A,
               const unsigned short* __restrict__ Bt,
               float* __restrict__ Co) {
    const int lane = threadIdx.x & 31;
    const int wave = threadIdx.x >> 5;
    const int half = lane >> 4;
    const int ln   = lane & 15;

    const int m0 = blockIdx.x * 128 + wave * 32;
    const int n0 = blockIdx.y * 64;

    const unsigned short* arow0 = A + (size_t)(m0 + ln) * CC;
    const unsigned short* arow1 = A + (size_t)(m0 + 16 + ln) * CC;
    const unsigned short* brow0 = Bt + (size_t)(n0 + 0  + ln) * CC;
    const unsigned short* brow1 = Bt + (size_t)(n0 + 16 + ln) * CC;
    const unsigned short* brow2 = Bt + (size_t)(n0 + 32 + ln) * CC;
    const unsigned short* brow3 = Bt + (size_t)(n0 + 48 + ln) * CC;

    v8f acc[2][4] = {};
    Frag16 a0[2], b0[4], a1[2], b1[4];

#define LDF(f, p, k)                                                          \
    do {                                                                      \
        (f).q[0] = *(const uint4*)((p) + (k) + half * 8);                     \
        (f).q[1] = *(const uint4*)((p) + (k) + 16 + half * 8);                \
    } while (0)

#define LDCHUNK(fa, fb, k)                                                    \
    do {                                                                      \
        LDF((fa)[0], arow0, (k));                                             \
        LDF((fa)[1], arow1, (k));                                             \
        LDF((fb)[0], brow0, (k));                                             \
        LDF((fb)[1], brow1, (k));                                             \
        LDF((fb)[2], brow2, (k));                                             \
        LDF((fb)[3], brow3, (k));                                             \
    } while (0)

#define WMMA8(fa, fb)                                                         \
    do {                                                                      \
        _Pragma("unroll") for (int j = 0; j < 4; ++j) {                       \
            acc[0][j] = __builtin_amdgcn_wmma_f32_16x16x32_bf16(              \
                false, (fa)[0].v, false, (fb)[j].v, (short)0, acc[0][j],      \
                false, false);                                                \
            acc[1][j] = __builtin_amdgcn_wmma_f32_16x16x32_bf16(              \
                false, (fa)[1].v, false, (fb)[j].v, (short)0, acc[1][j],      \
                false, false);                                                \
        }                                                                     \
    } while (0)

    // prime stage 0 with k = 0
    LDCHUNK(a0, b0, 0);

#pragma unroll 1
    for (int k = 0; k < CC; k += 64) {
        LDCHUNK(a1, b1, k + 32);        // prefetch odd chunk
        WMMA8(a0, b0);                  // compute even chunk
        if (k + 64 < CC)
            LDCHUNK(a0, b0, k + 64);    // prefetch next even chunk
        WMMA8(a1, b1);                  // compute odd chunk
    }
#undef WMMA8
#undef LDCHUNK
#undef LDF

    // C/D layout: VGPR i, lane -> row m = i + 8*half, col n = ln
#pragma unroll
    for (int r = 0; r < 2; ++r)
#pragma unroll
        for (int j = 0; j < 4; ++j)
#pragma unroll
            for (int i = 0; i < 8; ++i)
                Co[(size_t)(m0 + r * 16 + i + 8 * half) * CC + n0 + j * 16 +
                   ln] = acc[r][j][i];
}

// ---------------------------------------------------------------------------
// Kernel 4: RoPE + pack.
//   Qf,Kf,Vf are fp32 [B*T, C].  Outputs:
//     qb,kb bf16 [B,H,T,D]  (rows contiguous in D -> WMMA A/B frags for QK^T)
//     vt    bf16 [B,H,D,T]  (rows contiguous in T -> WMMA B frags for PV)
// ---------------------------------------------------------------------------
__global__ void rope_pack(const float* __restrict__ Qf,
                          const float* __restrict__ Kf,
                          const float* __restrict__ Vf,
                          unsigned short* __restrict__ qb,
                          unsigned short* __restrict__ kb,
                          unsigned short* __restrict__ vt) {
    int idx = blockIdx.x * blockDim.x + threadIdx.x;
    if (idx >= BB * TT * CC) return;
    int c  = idx % CC;
    int bt = idx / CC;
    int t  = bt % TT;
    int b  = bt / TT;
    int h  = c / DD;
    int d  = c & (DD - 1);
    int fi = d & 31;

    // inv_freq = 10000^(-fi/32)
    float inv = __expf(-(float)fi * (9.210340371976184f / 32.0f));
    float th  = (float)t * inv;
    float sn, cs;
    __sincosf(th, &sn, &cs);

    float qv = Qf[idx];
    float kv = Kf[idx];
    float qo = (d < 32) ? -Qf[idx + 32] : Qf[idx - 32];
    float ko = (d < 32) ? -Kf[idx + 32] : Kf[idx - 32];

    size_t oidx = (((size_t)(b * HH + h) * TT) + t) * DD + d;
    qb[oidx] = f2bf(qv * cs + qo * sn);
    kb[oidx] = f2bf(kv * cs + ko * sn);
    vt[(((size_t)(b * HH + h) * DD) + d) * TT + t] = f2bf(Vf[idx]);
}

// ---------------------------------------------------------------------------
// Kernel 5: causal flash attention, one wave per (b,h, 16-query tile).
//   S = (Q K^T)/8  -> streaming softmax -> O += P V ; O /= l
//   Output attnb: bf16 [B*T, C]  (row = b*T + q, col = h*64 + d)
// ---------------------------------------------------------------------------
__global__ void flash_attn_wmma(const unsigned short* __restrict__ qb,
                                const unsigned short* __restrict__ kb,
                                const unsigned short* __restrict__ vt,
                                unsigned short* __restrict__ attnb) {
    __shared__ __align__(16) unsigned short plds[4][16 * 32];

    const int lane = threadIdx.x & 31;
    const int wave = threadIdx.x >> 5;
    const int half = lane >> 4;
    const int ln   = lane & 15;

    const int g  = blockIdx.x * 4 + wave;     // global wave id
    const int qt = g & 127;                   // T/16 = 128 query tiles
    const int bh = g >> 7;                    // 0 .. B*H-1
    const int q0 = qt * 16;

    const unsigned short* qbase = qb + (size_t)bh * TT * DD;
    const unsigned short* kbase = kb + (size_t)bh * TT * DD;
    const unsigned short* vbase = vt + (size_t)bh * DD * TT;

    // Q A-fragments for d-windows [0,32) and [32,64) (fixed for this tile)
    const unsigned short* qrow = qbase + (size_t)(q0 + ln) * DD;
    Frag16 aq0, aq1;
    aq0.q[0] = *(const uint4*)(qrow + 0  + half * 8);
    aq0.q[1] = *(const uint4*)(qrow + 16 + half * 8);
    aq1.q[0] = *(const uint4*)(qrow + 32 + half * 8);
    aq1.q[1] = *(const uint4*)(qrow + 48 + half * 8);

    float mi[8], li[8];
#pragma unroll
    for (int i = 0; i < 8; ++i) { mi[i] = -3.0e38f; li[i] = 0.0f; }
    v8f acc[4] = {};

    const int nkb = (q0 + 47) >> 5;           // 32-key blocks covering 0..q0+15
#pragma unroll 1
    for (int kbk = 0; kbk < nkb; ++kbk) {
        const int j0 = kbk * 32;

        // ---- S = Q K^T for 16 queries x 32 keys (two 16x16 C tiles) ----
        v8f s[2];
#pragma unroll
        for (int sub = 0; sub < 2; ++sub) {
            const unsigned short* krow =
                kbase + (size_t)(j0 + sub * 16 + ln) * DD;
            Frag16 b0, b1;
            b0.q[0] = *(const uint4*)(krow + 0  + half * 8);
            b0.q[1] = *(const uint4*)(krow + 16 + half * 8);
            b1.q[0] = *(const uint4*)(krow + 32 + half * 8);
            b1.q[1] = *(const uint4*)(krow + 48 + half * 8);
            v8f z = {};
            z = __builtin_amdgcn_wmma_f32_16x16x32_bf16(
                false, aq0.v, false, b0.v, (short)0, z, false, false);
            z = __builtin_amdgcn_wmma_f32_16x16x32_bf16(
                false, aq1.v, false, b1.v, (short)0, z, false, false);
            s[sub] = z;
        }

        // ---- scale, causal mask, streaming softmax update ----
        float rmax[8];
#pragma unroll
        for (int i = 0; i < 8; ++i) {
            const int qrowg = q0 + i + 8 * half;
#pragma unroll
            for (int sub = 0; sub < 2; ++sub) {
                int col = j0 + sub * 16 + ln;
                float v = s[sub][i] * 0.125f;         // 1/sqrt(64)
                if (col > qrowg) v = NEG_INF_F;
                s[sub][i] = v;
            }
            float t = fmaxf(s[0][i], s[1][i]);
            for (int off = 1; off < 16; off <<= 1)
                t = fmaxf(t, __shfl_xor(t, off, 32));
            rmax[i] = t;
        }
#pragma unroll
        for (int i = 0; i < 8; ++i) {
            float mnew = fmaxf(mi[i], rmax[i]);
            float sc   = __expf(mi[i] - mnew);
            mi[i] = mnew;
            float p0 = __expf(s[0][i] - mnew);
            float p1 = __expf(s[1][i] - mnew);
            s[0][i] = p0;
            s[1][i] = p1;
            float rs = p0 + p1;
            for (int off = 1; off < 16; off <<= 1)
                rs += __shfl_xor(rs, off, 32);
            li[i] = li[i] * sc + rs;
#pragma unroll
            for (int j = 0; j < 4; ++j) acc[j][i] *= sc;
        }

        // ---- transpose P (C-layout) into A-layout via per-wave LDS ----
#pragma unroll
        for (int sub = 0; sub < 2; ++sub)
#pragma unroll
            for (int i = 0; i < 8; ++i)
                plds[wave][(i + 8 * half) * 32 + sub * 16 + ln] =
                    f2bf(s[sub][i]);
        asm volatile("s_wait_dscnt 0" ::: "memory");

        const unsigned short* prow = &plds[wave][ln * 32];
        Frag16 pa;
        pa.q[0] = *(const uint4*)(prow + 0  + half * 8);
        pa.q[1] = *(const uint4*)(prow + 16 + half * 8);

        // ---- O += P V  (V^T rows contiguous over key index) ----
#pragma unroll
        for (int j = 0; j < 4; ++j) {
            const unsigned short* vrow =
                vbase + (size_t)(j * 16 + ln) * TT + j0;
            Frag16 bv;
            bv.q[0] = *(const uint4*)(vrow + 0  + half * 8);
            bv.q[1] = *(const uint4*)(vrow + 16 + half * 8);
            acc[j] = __builtin_amdgcn_wmma_f32_16x16x32_bf16(
                false, pa.v, false, bv.v, (short)0, acc[j], false, false);
        }
    }

    // ---- epilogue: O /= l, write bf16 into [B*T, C] ----
    const int b = bh / HH;
    const int h = bh - b * HH;
#pragma unroll
    for (int j = 0; j < 4; ++j)
#pragma unroll
        for (int i = 0; i < 8; ++i) {
            float o = acc[j][i] / li[i];
            int row = q0 + i + 8 * half;
            attnb[((size_t)(b * TT + row)) * CC + h * DD + j * 16 + ln] =
                f2bf(o);
        }
}

// ---------------------------------------------------------------------------
// Host launcher
// ---------------------------------------------------------------------------
extern "C" void kernel_launch(void* const* d_in, const int* in_sizes, int n_in,
                              void* d_out, int out_size, void* d_ws,
                              size_t ws_size, hipStream_t stream) {
    (void)in_sizes; (void)n_in; (void)out_size; (void)ws_size;
    const float* x  = (const float*)d_in[0];
    const float* wq = (const float*)d_in[1];
    const float* wk = (const float*)d_in[2];
    const float* wv = (const float*)d_in[3];
    const float* wo = (const float*)d_in[4];
    // d_in[5] = mask (unused; causal mask applied analytically)

    const size_t nx = (size_t)MM * CC;

    char* ws = (char*)d_ws;
    size_t o = 0;
    auto alloc = [&](size_t bytes) { char* p = ws + o; o += (bytes + 255) & ~(size_t)255; return p; };

    unsigned short* xb   = (unsigned short*)alloc(nx * 2);
    unsigned short* wqT  = (unsigned short*)alloc((size_t)CC * CC * 2);
    unsigned short* wkT  = (unsigned short*)alloc((size_t)CC * CC * 2);
    unsigned short* wvT  = (unsigned short*)alloc((size_t)CC * CC * 2);
    unsigned short* woT  = (unsigned short*)alloc((size_t)CC * CC * 2);
    float*          Qf   = (float*)alloc(nx * 4);
    float*          Kf   = (float*)alloc(nx * 4);
    float*          Vf   = (float*)alloc(nx * 4);
    unsigned short* qbp  = (unsigned short*)alloc(nx * 2);
    unsigned short* kbp  = (unsigned short*)alloc(nx * 2);
    unsigned short* vtp  = (unsigned short*)alloc(nx * 2);
    unsigned short* attn = (unsigned short*)alloc(nx * 2);

    // 1) convert activations
    cvt_f32_bf16<<<dim3((unsigned)((nx + 255) / 256)), dim3(256), 0, stream>>>(
        x, xb, (int)nx);

    // 2) transpose+convert weights
    const int nw = CC * CC;
    transpose_w_bf16<<<dim3((nw + 255) / 256), dim3(256), 0, stream>>>(wq, wqT);
    transpose_w_bf16<<<dim3((nw + 255) / 256), dim3(256), 0, stream>>>(wk, wkT);
    transpose_w_bf16<<<dim3((nw + 255) / 256), dim3(256), 0, stream>>>(wv, wvT);
    transpose_w_bf16<<<dim3((nw + 255) / 256), dim3(256), 0, stream>>>(wo, woT);

    // 3) QKV projections (WMMA GEMMs): grid(M/128, N/64), 4 waves/block
    dim3 ggrid(MM / 128, CC / 64), gblk(128);
    gemm_bf16_wmma<<<ggrid, gblk, 0, stream>>>(xb, wqT, Qf);
    gemm_bf16_wmma<<<ggrid, gblk, 0, stream>>>(xb, wkT, Kf);
    gemm_bf16_wmma<<<ggrid, gblk, 0, stream>>>(xb, wvT, Vf);

    // 4) RoPE + layout pack
    rope_pack<<<dim3((unsigned)((nx + 255) / 256)), dim3(256), 0, stream>>>(
        Qf, Kf, Vf, qbp, kbp, vtp);

    // 5) flash attention: B*H*(T/16) waves, 4 waves per block
    const int nwaves = BB * HH * (TT / 16);          // 6144
    flash_attn_wmma<<<dim3(nwaves / 4), dim3(128), 0, stream>>>(qbp, kbp, vtp,
                                                                attn);

    // 6) output projection -> fp32 d_out
    gemm_bf16_wmma<<<ggrid, gblk, 0, stream>>>(attn, woT, (float*)d_out);
}
